// Flux2ParallelSelfAttention_83236466196677
// MI455X (gfx1250) — matrix operand order repaired and checked by reading, not compiled, not run on hardware
//
#include <hip/hip_runtime.h>
#include <hip/hip_bf16.h>

typedef __bf16 bf16;
typedef __attribute__((ext_vector_type(16))) __bf16 v16bf;
typedef __attribute__((ext_vector_type(8)))  float  v8f;

#define Q_DIM     2048
#define HEADS     16
#define HEAD_DIM  128
#define INNER     2048
#define MLP_HID   8192
#define FUSED_OUT 22528      // 3*INNER + 2*MLP_HID
#define CAT_DIM   10240      // INNER + MLP_HID
#define BATCH     2
#define SEQ       2048
#define ROWS      4096       // BATCH*SEQ
#define ATTN_SCALE 0.08838834764831845f  // 1/sqrt(128)

// ---------------------------------------------------------------------------
// CDNA5 async global->LDS copy (ASYNCcnt-tracked), via inline asm so it is
// portable across ROCm/upstream toolchains. Each lane copies 16 bytes from
// its global address directly into LDS at its 32-bit LDS byte offset.
// Generic (flat) pointers to __shared__ carry the LDS offset in bits [31:0].
// ---------------------------------------------------------------------------
__device__ inline unsigned lds_off(const void* p) {
    return (unsigned)(unsigned long long)(uintptr_t)p;
}
__device__ inline void async_copy16(unsigned ldsAddr, const void* gptr) {
    asm volatile("global_load_async_to_lds_b128 %0, %1, off"
                 :: "v"(ldsAddr), "v"((unsigned long long)(uintptr_t)gptr)
                 : "memory");
}
__device__ inline void wait_async0() {
    asm volatile("s_wait_asynccnt 0x0" ::: "memory");
}

// ---------------------------------------------------------------------------
// WMMA fragment helpers (CDNA5 16x16x32 bf16 layouts, wave32)
// A (16x32): lane L -> row m=L%16, half=L/16; VGPR0-3: K=half*8+0..7,
//            VGPR4-7: K=16+half*8+0..7  => two b128 loads.
// B (32x16): lane L -> col n=L%16, half=L/16; VGPR0-3: K=half*8+0..7,
//            VGPR4-7: K=16+half*8+0..7  => two b128 loads.
// C/D (16x16 f32, 8 VGPR): elem v -> row m=v+8*(L/16), col n=L%16.
// ---------------------------------------------------------------------------
__device__ inline v16bf frag_A(const bf16* row, int hf, int kOff) {
    union { v16bf v; uint4 q[2]; } u;
    u.q[0] = *(const uint4*)(row + kOff + hf * 8);
    u.q[1] = *(const uint4*)(row + kOff + 16 + hf * 8);
    return u.v;
}
__device__ inline v16bf frag_B(const bf16* row, int hf, int kOff) {
    union { v16bf v; uint4 q[2]; } u;
    u.q[0] = *(const uint4*)(row + kOff + hf * 8);
    u.q[1] = *(const uint4*)(row + kOff + 16 + hf * 8);
    return u.v;
}
__device__ inline v8f wmma_bf(v16bf a, v16bf b, v8f c) {
    return __builtin_amdgcn_wmma_f32_16x16x32_bf16(
        false, a, false, b, (short)0, c, false, false);
}

// ---------------------------------------------------------------------------
// fp32 -> bf16 conversion (grid-stride)
// ---------------------------------------------------------------------------
__global__ __launch_bounds__(256) void cvt_f32_bf16(
    const float* __restrict__ src, bf16* __restrict__ dst, long n) {
    long i = (long)blockIdx.x * blockDim.x + threadIdx.x;
    long stride = (long)gridDim.x * blockDim.x;
    for (; i < n; i += stride) dst[i] = (bf16)src[i];
}

// ---------------------------------------------------------------------------
// One-time weight repack: fp32 W[K][N] -> bf16 W^T[N][K].
// ---------------------------------------------------------------------------
__global__ __launch_bounds__(256) void transpose_cvt(
    const float* __restrict__ src, bf16* __restrict__ dst, int K, int N) {
    __shared__ float tile[32][33];
    const int tx = threadIdx.x & 31;
    const int ty = threadIdx.x >> 5;
    const int n0 = blockIdx.x * 32, k0 = blockIdx.y * 32;
    for (int r = ty; r < 32; r += 8)
        tile[r][tx] = src[(long)(k0 + r) * N + n0 + tx];
    __syncthreads();
    for (int r = ty; r < 32; r += 8)
        dst[(long)(n0 + r) * K + k0 + tx] = (bf16)tile[tx][r];
}

// ---------------------------------------------------------------------------
// bf16 WMMA GEMM:  C[M,N](f32) = A[M,K] * Bt[N,K]^T  (+ bias)
// Double-buffered LDS fed by async global->LDS copies (ASYNCcnt), so the
// next tile streams into one buffer while WMMAs consume the other.
// Block tile 128x128, 8 waves (2x4), wave tile 64x32, BK=32.
// ---------------------------------------------------------------------------
#define GBM 128
#define GBN 128
#define GBK 32
#define LDA 40   // LDS row stride (bf16): b128-aligned, spreads banks

template <bool ADD_BIAS>
__global__ __launch_bounds__(256) void gemm_bf16(
    const bf16* __restrict__ A, const bf16* __restrict__ Bt,
    float* __restrict__ C, const float* __restrict__ bias,
    int M, int N, int K) {
    __shared__ bf16 As[2][GBM * LDA];
    __shared__ bf16 Bs[2][GBN * LDA];  // Bs[n][k]

    const int t = threadIdx.x;
    const int lane = t & 31, wave = t >> 5;
    const int hf = lane >> 4, ln = lane & 15;
    const int m0 = blockIdx.y * GBM, n0 = blockIdx.x * GBN;
    const int wm0 = (wave >> 2) * 64;  // 2 waves in M
    const int wn0 = (wave & 3) * 32;   // 4 waves in N

    v8f acc[4][2];
    for (int mi = 0; mi < 4; ++mi)
        for (int ni = 0; ni < 2; ++ni)
            for (int e = 0; e < 8; ++e) acc[mi][ni][e] = 0.0f;

    const int srow = t >> 1, sseg = t & 1;  // 128 rows x 2 segs of 16
    const bf16* aSrc = A + (long)(m0 + srow) * K + sseg * 16;
    const bf16* bSrc = Bt + (long)(n0 + srow) * K + sseg * 16;
    const int ldsIdx = srow * LDA + sseg * 16;
    const unsigned aOff[2] = { lds_off(&As[0][ldsIdx]), lds_off(&As[1][ldsIdx]) };
    const unsigned bOff[2] = { lds_off(&Bs[0][ldsIdx]), lds_off(&Bs[1][ldsIdx]) };

    // prologue: async-stage tile 0 into buffer 0
    async_copy16(aOff[0], aSrc);
    async_copy16(aOff[0] + 16, aSrc + 8);
    async_copy16(bOff[0], bSrc);
    async_copy16(bOff[0] + 16, bSrc + 8);

    int buf = 0;
    for (int k0 = 0; k0 < K; k0 += GBK, buf ^= 1) {
        wait_async0();       // own async writes of tile k0 complete
        __syncthreads();     // all waves' writes visible; prev compute done

        if (k0 + GBK < K) {  // stream next tile into the other buffer
            const bf16* ag = aSrc + k0 + GBK;
            const bf16* bg = bSrc + k0 + GBK;
            async_copy16(aOff[buf ^ 1], ag);
            async_copy16(aOff[buf ^ 1] + 16, ag + 8);
            async_copy16(bOff[buf ^ 1], bg);
            async_copy16(bOff[buf ^ 1] + 16, bg + 8);
        }

        v16bf af[4], bfv[2];
        for (int mi = 0; mi < 4; ++mi)
            af[mi] = frag_A(&As[buf][(wm0 + mi * 16 + ln) * LDA], hf, 0);
        for (int ni = 0; ni < 2; ++ni)
            bfv[ni] = frag_B(&Bs[buf][(wn0 + ni * 16 + ln) * LDA], hf, 0);
        for (int mi = 0; mi < 4; ++mi)
            for (int ni = 0; ni < 2; ++ni)
                acc[mi][ni] = wmma_bf(af[mi], bfv[ni], acc[mi][ni]);
    }

    for (int mi = 0; mi < 4; ++mi)
        for (int ni = 0; ni < 2; ++ni) {
            const int col = n0 + wn0 + ni * 16 + ln;
            const float bv = ADD_BIAS ? bias[col] : 0.0f;
            for (int v = 0; v < 8; ++v) {
                const int row = m0 + wm0 + mi * 16 + v + 8 * hf;
                C[(long)row * N + col] = acc[mi][ni][v] + bv;
            }
        }
}

// ---------------------------------------------------------------------------
// RMSNorm(q,k) + interleaved RoPE + fold attn scale into q.
// Writes q/k bf16 [B,H,S,D]; V transposed [B,H,D,S].
// ---------------------------------------------------------------------------
__global__ __launch_bounds__(256) void qkv_norm_rope(
    const float* __restrict__ proj, const float* __restrict__ cosT,
    const float* __restrict__ sinT, const float* __restrict__ wq,
    const float* __restrict__ wk, bf16* __restrict__ Qd,
    bf16* __restrict__ Kd, bf16* __restrict__ Vt) {
    const int lane = threadIdx.x & 31, wave = threadIdx.x >> 5;
    const long gw = (long)blockIdx.x * 8 + wave;
    const int h = (int)(gw % HEADS);
    const long row = gw / HEADS;
    const int s = (int)(row % SEQ);
    const long b = row / SEQ;
    const int d0 = lane * 4;

    const float* base = proj + row * (long)FUSED_OUT;
    float q[4], k[4], v[4];
    for (int i = 0; i < 4; ++i) {
        q[i] = base[0 * INNER + h * HEAD_DIM + d0 + i];
        k[i] = base[1 * INNER + h * HEAD_DIM + d0 + i];
        v[i] = base[2 * INNER + h * HEAD_DIM + d0 + i];
    }
    float sq = 0.f, sk = 0.f;
    for (int i = 0; i < 4; ++i) { sq += q[i] * q[i]; sk += k[i] * k[i]; }
    for (int m = 1; m < 32; m <<= 1) {
        sq += __shfl_xor(sq, m, 32);
        sk += __shfl_xor(sk, m, 32);
    }
    const float rq = rsqrtf(sq * (1.0f / HEAD_DIM) + 1e-5f);
    const float rk = rsqrtf(sk * (1.0f / HEAD_DIM) + 1e-5f);
    for (int i = 0; i < 4; ++i) {
        q[i] *= rq * wq[d0 + i];
        k[i] *= rk * wk[d0 + i];
    }
    const float* cr = cosT + (long)s * HEAD_DIM + d0;
    const float* sr = sinT + (long)s * HEAD_DIM + d0;
    for (int p = 0; p < 2; ++p) {
        const float c0 = cr[2 * p], c1 = cr[2 * p + 1];
        const float s0 = sr[2 * p], s1 = sr[2 * p + 1];
        float x0 = q[2 * p], x1 = q[2 * p + 1];
        q[2 * p] = x0 * c0 - x1 * s0;
        q[2 * p + 1] = x1 * c1 + x0 * s1;
        x0 = k[2 * p]; x1 = k[2 * p + 1];
        k[2 * p] = x0 * c0 - x1 * s0;
        k[2 * p + 1] = x1 * c1 + x0 * s1;
    }
    const long o = (((b * HEADS + h) * SEQ) + s) * (long)HEAD_DIM + d0;
    const long vtBase = ((b * HEADS + h) * HEAD_DIM + d0) * (long)SEQ + s;
    for (int i = 0; i < 4; ++i) {
        Qd[o + i] = (bf16)(q[i] * ATTN_SCALE);
        Kd[o + i] = (bf16)k[i];
        Vt[vtBase + (long)i * SEQ] = (bf16)v[i];
    }
}

// ---------------------------------------------------------------------------
// Flash-style attention, WMMA bf16, async double-buffered K/V^T staging.
// Block: 8 waves x 16 q-rows = 128 q-rows per (b,h). Key blocks of 64.
// ---------------------------------------------------------------------------
#define KB  64
#define LKT 136   // K-tile LDS stride  (row-major [key][d])
#define LVT 72    // V^T / P LDS stride

__global__ __launch_bounds__(256) void attention_wmma(
    const bf16* __restrict__ Q, const bf16* __restrict__ Kd,
    const bf16* __restrict__ VtG, bf16* __restrict__ Out) {
    __shared__ bf16 Kt[2][KB * LKT];           // [key][d]
    __shared__ bf16 Vt[2][HEAD_DIM * LVT];     // [d][key]
    __shared__ bf16 Pb[8 * 16 * LVT];          // per-wave P staging

    const int t = threadIdx.x, lane = t & 31, wave = t >> 5;
    const int hf = lane >> 4, ln = lane & 15;
    const int b = blockIdx.z, h = blockIdx.y;
    const int q0 = blockIdx.x * 128 + wave * 16;

    const long headOff = (((long)b * HEADS + h) * SEQ) * HEAD_DIM;
    const bf16* qbase = Q + headOff;
    const bf16* kbase = Kd + headOff;
    const bf16* vtbase = VtG + headOff;  // [d][s] within this head

    // staging geometry
    const int key = t & 63, dseg = t >> 6;  // K: 64 keys x 4 segs of 32d
    const int vdr = t >> 1, vseg = t & 1;   // V: 128 d-rows x 2 segs of 32 keys
    const bf16* kSrc = kbase + (long)key * HEAD_DIM + dseg * 32;
    const bf16* vSrc = vtbase + (long)vdr * SEQ + vseg * 32;
    const int kIdx = key * LKT + dseg * 32;
    const int vIdx = vdr * LVT + vseg * 32;
    const unsigned kOffL[2] = { lds_off(&Kt[0][kIdx]), lds_off(&Kt[1][kIdx]) };
    const unsigned vOffL[2] = { lds_off(&Vt[0][vIdx]), lds_off(&Vt[1][vIdx]) };

    // Q fragments resident: 16 rows x 128 d
    v16bf qf[4];
    for (int dk = 0; dk < 4; ++dk)
        qf[dk] = frag_A(qbase + (long)(q0 + ln) * HEAD_DIM, hf, dk * 32);

    v8f o[8];
    for (int dt = 0; dt < 8; ++dt)
        for (int e = 0; e < 8; ++e) o[dt][e] = 0.0f;
    float mrow[8], lrow[8];
    for (int v = 0; v < 8; ++v) { mrow[v] = -1e30f; lrow[v] = 0.0f; }

    bf16* pb = &Pb[wave * 16 * LVT];

    // prologue: async-stage first key block
    for (int c = 0; c < 4; ++c)
        async_copy16(kOffL[0] + c * 16, kSrc + c * 8);
    for (int c = 0; c < 4; ++c)
        async_copy16(vOffL[0] + c * 16, vSrc + c * 8);

    int kbuf = 0;
    for (int kb = 0; kb < SEQ; kb += KB, kbuf ^= 1) {
        wait_async0();
        __syncthreads();

        if (kb + KB < SEQ) {  // stream next key block into the other buffer
            const bf16* kg = kSrc + (long)(kb + KB) * HEAD_DIM;
            const bf16* vg = vSrc + (kb + KB);
            for (int c = 0; c < 4; ++c)
                async_copy16(kOffL[kbuf ^ 1] + c * 16, kg + c * 8);
            for (int c = 0; c < 4; ++c)
                async_copy16(vOffL[kbuf ^ 1] + c * 16, vg + c * 8);
        }

        // ---- scores: S(16q x 64k) = Q x K^T ----
        v8f sc[4];
        for (int nt = 0; nt < 4; ++nt) {
            for (int e = 0; e < 8; ++e) sc[nt][e] = 0.0f;
            for (int dk = 0; dk < 4; ++dk)
                sc[nt] = wmma_bf(qf[dk],
                                 frag_B(&Kt[kbuf][(nt * 16 + ln) * LKT], hf, dk * 32),
                                 sc[nt]);
        }

        // ---- online softmax (row stats across 16-lane n-groups) ----
        float mnew[8], rs[8], corr[8];
        for (int v = 0; v < 8; ++v) {
            float mx = sc[0][v];
            for (int nt = 1; nt < 4; ++nt) mx = fmaxf(mx, sc[nt][v]);
            for (int m = 1; m < 16; m <<= 1) mx = fmaxf(mx, __shfl_xor(mx, m, 32));
            mnew[v] = fmaxf(mrow[v], mx);
        }
        for (int v = 0; v < 8; ++v) {
            float r = 0.0f;
            for (int nt = 0; nt < 4; ++nt) {
                float p = __expf(sc[nt][v] - mnew[v]);
                sc[nt][v] = p;
                r += p;
            }
            rs[v] = r;
        }
        for (int v = 0; v < 8; ++v)
            for (int m = 1; m < 16; m <<= 1) rs[v] += __shfl_xor(rs[v], m, 32);
        for (int v = 0; v < 8; ++v) {
            corr[v] = __expf(mrow[v] - mnew[v]);
            lrow[v] = lrow[v] * corr[v] + rs[v];
            mrow[v] = mnew[v];
        }
        for (int dt = 0; dt < 8; ++dt)
            for (int v = 0; v < 8; ++v) o[dt][v] *= corr[v];

        // ---- re-layout P (C-layout -> A-layout) through per-wave LDS ----
        for (int nt = 0; nt < 4; ++nt)
            for (int v = 0; v < 8; ++v)
                pb[(v + 8 * hf) * LVT + nt * 16 + ln] = (bf16)sc[nt][v];

        v16bf pf[2];
        for (int kt = 0; kt < 2; ++kt)
            pf[kt] = frag_A(pb + ln * LVT, hf, kt * 32);

        // ---- O += P x V ----
        for (int dt = 0; dt < 8; ++dt)
            for (int kt = 0; kt < 2; ++kt)
                o[dt] = wmma_bf(pf[kt],
                                frag_B(&Vt[kbuf][(dt * 16 + ln) * LVT], hf, kt * 32),
                                o[dt]);
    }

    // ---- epilogue: O /= l, write bf16 into attn part of cat buffer ----
    const long rowBase = (long)b * SEQ + q0;
    for (int dt = 0; dt < 8; ++dt)
        for (int v = 0; v < 8; ++v) {
            const long r = rowBase + v + 8 * hf;
            Out[r * CAT_DIM + h * HEAD_DIM + dt * 16 + ln] =
                (bf16)(o[dt][v] / lrow[v]);
        }
}

// ---------------------------------------------------------------------------
// SwiGLU: silu(m1) * m2 -> bf16 into mlp part of cat buffer
// ---------------------------------------------------------------------------
__global__ __launch_bounds__(256) void swiglu(
    const float* __restrict__ proj, bf16* __restrict__ cat) {
    const long n = (long)ROWS * MLP_HID;
    long i = (long)blockIdx.x * blockDim.x + threadIdx.x;
    const long stride = (long)gridDim.x * blockDim.x;
    for (; i < n; i += stride) {
        const long row = i / MLP_HID, j = i % MLP_HID;
        const float a = proj[row * (long)FUSED_OUT + 3 * INNER + j];
        const float g = proj[row * (long)FUSED_OUT + 3 * INNER + MLP_HID + j];
        const float s = a / (1.0f + __expf(-a));
        cat[row * (long)CAT_DIM + INNER + j] = (bf16)(s * g);
    }
}

// ---------------------------------------------------------------------------
// Host launcher
// ---------------------------------------------------------------------------
extern "C" void kernel_launch(void* const* d_in, const int* in_sizes, int n_in,
                              void* d_out, int out_size, void* d_ws, size_t ws_size,
                              hipStream_t stream) {
    const float* hidden = (const float*)d_in[0];
    const float* rope_cos = (const float*)d_in[1];
    const float* rope_sin = (const float*)d_in[2];
    const float* W_qkv = (const float*)d_in[3];
    const float* norm_q = (const float*)d_in[4];
    const float* norm_k = (const float*)d_in[5];
    const float* W_out = (const float*)d_in[6];
    const float* b_out = (const float*)d_in[7];

    char* ws = (char*)d_ws;
    size_t off = 0;
    auto take = [&](size_t bytes) -> char* {
        char* p = ws + off;
        off = (off + bytes + 255) & ~(size_t)255;
        return p;
    };
    bf16* hsb   = (bf16*)take((size_t)ROWS * Q_DIM * sizeof(bf16));
    bf16* wqbT  = (bf16*)take((size_t)Q_DIM * FUSED_OUT * sizeof(bf16));   // [22528][2048]
    bf16* wobT  = (bf16*)take((size_t)CAT_DIM * Q_DIM * sizeof(bf16));     // [2048][10240]
    float* proj = (float*)take((size_t)ROWS * FUSED_OUT * sizeof(float));
    bf16* qb    = (bf16*)take((size_t)ROWS * INNER * sizeof(bf16));
    bf16* kb    = (bf16*)take((size_t)ROWS * INNER * sizeof(bf16));
    bf16* vtb   = (bf16*)take((size_t)ROWS * INNER * sizeof(bf16));        // [B,H,D,S]
    bf16* cat   = (bf16*)take((size_t)ROWS * CAT_DIM * sizeof(bf16));
    (void)ws_size; (void)in_sizes; (void)n_in; (void)out_size;

    // 1) convert activations; transpose+convert weights (paid once)
    cvt_f32_bf16<<<2048, 256, 0, stream>>>(hidden, hsb, (long)ROWS * Q_DIM);
    {
        dim3 g1(FUSED_OUT / 32, Q_DIM / 32);
        transpose_cvt<<<g1, 256, 0, stream>>>(W_qkv, wqbT, Q_DIM, FUSED_OUT);
        dim3 g2(Q_DIM / 32, CAT_DIM / 32);
        transpose_cvt<<<g2, 256, 0, stream>>>(W_out, wobT, CAT_DIM, Q_DIM);
    }

    // 2) GEMM1: proj = hs @ W_qkv_mlp   [4096 x 22528]
    {
        dim3 grid(FUSED_OUT / GBN, ROWS / GBM);
        gemm_bf16<false><<<grid, 256, 0, stream>>>(hsb, wqbT, proj, nullptr,
                                                   ROWS, FUSED_OUT, Q_DIM);
    }

    // 3) RMSNorm + RoPE -> q/k bf16 head-major, V^T
    qkv_norm_rope<<<(ROWS * HEADS) / 8, 256, 0, stream>>>(
        proj, rope_cos, rope_sin, norm_q, norm_k, qb, kb, vtb);

    // 4) attention -> attn columns of cat buffer
    {
        dim3 grid(SEQ / 128, HEADS, BATCH);
        attention_wmma<<<grid, 256, 0, stream>>>(qb, kb, vtb, cat);
    }

    // 5) SwiGLU -> mlp columns of cat buffer
    swiglu<<<4096, 256, 0, stream>>>(proj, cat);

    // 6) GEMM2: out = cat @ W_out + b_out   [4096 x 2048], fp32 output
    {
        dim3 grid(Q_DIM / GBN, ROWS / GBM);
        gemm_bf16<true><<<grid, 256, 0, stream>>>(cat, wobT, (float*)d_out, b_out,
                                                  ROWS, Q_DIM, CAT_DIM);
    }
}
